// EncoderLayer_9405978378868
// MI455X (gfx1250) — 16-bit and 8-bit matrix operand order repaired and checked by reading, compiled, not run on hardware
//
#include <hip/hip_runtime.h>
#include <cstdint>
#include <cstddef>

#define BATCH   2
#define SEQ     1024
#define DMODEL  1024
#define DINNER  2048
#define DSTATE  16
#define DTRANK  64
#define DFF     4096
#define NTOK    (BATCH*SEQ)

typedef __attribute__((ext_vector_type(16))) __bf16 v16bf;
typedef __attribute__((ext_vector_type(8)))  float  v8f;
typedef __attribute__((ext_vector_type(8)))  int    v8i;

__device__ __forceinline__ unsigned short f2bf(float f) {
  unsigned u = __float_as_uint(f);
  u += 0x7FFFu + ((u >> 16) & 1u);
  return (unsigned short)(u >> 16);
}

__global__ void k_f32_to_bf16(const float* __restrict__ in,
                              unsigned short* __restrict__ out, int n) {
  int i = blockIdx.x * 256 + threadIdx.x;
  if (i < n) out[i] = f2bf(in[i]);
}

__device__ __forceinline__ v16bf ld_afrag_bf(const unsigned short* __restrict__ row,
                                             int k0, int half) {
  union { v16bf v; uint4 q[2]; } u;
  const unsigned short* p = row + k0 + half * 8;
  u.q[0] = *(const uint4*)(p);
  u.q[1] = *(const uint4*)(p + 16);
  return u.v;
}
__device__ __forceinline__ v16bf ld_bfrag_bf(const unsigned short* __restrict__ row,
                                             int k0, int half) {
  union { v16bf v; uint4 q[2]; } u;
  const unsigned short* p = row + k0 + half * 8;
  u.q[0] = *(const uint4*)(p);
  u.q[1] = *(const uint4*)(p + 16);
  return u.v;
}
__device__ __forceinline__ v8i ld_afrag_i8(const signed char* __restrict__ row,
                                           int k0, int half) {
  union { v8i v; uint4 q[2]; } u;
  const signed char* p = row + k0 + half * 16;
  u.q[0] = *(const uint4*)(p);
  u.q[1] = *(const uint4*)(p + 32);
  return u.v;
}
__device__ __forceinline__ v8i ld_bfrag_i8(const signed char* __restrict__ row,
                                           int k0, int half) {
  union { v8i v; uint4 q[2]; } u;
  const signed char* p = row + k0 + half * 16;
  u.q[0] = *(const uint4*)(p);
  u.q[1] = *(const uint4*)(p + 32);
  return u.v;
}

__global__ void k_gemm_bf16_blk(const unsigned short* __restrict__ A,
                                const unsigned short* __restrict__ W,
                                float* __restrict__ C, int M, int N, int K) {
  const int wave = threadIdx.x >> 5;
  const int lane = threadIdx.x & 31;
  const int mt = blockIdx.x * 4 + wave;
  const int tn = N >> 5;
  const int tm = mt / tn;
  const int tc = mt % tn;
  if (tm >= (M >> 6)) return;
  const int half = lane >> 4;
  const int r16  = lane & 15;
  const size_t m0 = (size_t)tm * 64, n0 = (size_t)tc * 32;
  const unsigned short* ar[4];
#pragma unroll
  for (int i = 0; i < 4; ++i) ar[i] = A + (m0 + i * 16 + r16) * (size_t)K;
  const unsigned short* br0 = W + (n0 + r16) * (size_t)K;
  const unsigned short* br1 = W + (n0 + 16 + r16) * (size_t)K;
  v8f acc[4][2] = {};
  for (int k0 = 0; k0 < K; k0 += 32) {
    v16bf b0 = ld_bfrag_bf(br0, k0, half);
    v16bf b1 = ld_bfrag_bf(br1, k0, half);
#pragma unroll
    for (int i = 0; i < 4; ++i) {
      v16bf a = ld_afrag_bf(ar[i], k0, half);
      acc[i][0] = __builtin_amdgcn_wmma_f32_16x16x32_bf16(false, a, false, b0,
                                                          (short)0, acc[i][0], false, false);
      acc[i][1] = __builtin_amdgcn_wmma_f32_16x16x32_bf16(false, a, false, b1,
                                                          (short)0, acc[i][1], false, false);
    }
  }
#pragma unroll
  for (int i = 0; i < 4; ++i)
#pragma unroll
    for (int j = 0; j < 2; ++j) {
      float* crow = C + (m0 + i * 16 + half * 8) * (size_t)N + n0 + j * 16 + r16;
#pragma unroll
      for (int r = 0; r < 8; ++r) crow[(size_t)r * N] = acc[i][j][r];
    }
}

__global__ void k_gemm_iu8_blk(const signed char* __restrict__ Aq,
                               const signed char* __restrict__ Wq,
                               const float* __restrict__ xs,
                               const float* __restrict__ wsv,
                               float* __restrict__ C, int M, int N, int K) {
  const int wave = threadIdx.x >> 5;
  const int lane = threadIdx.x & 31;
  const int mt = blockIdx.x * 4 + wave;
  const int tn = N >> 5;
  const int tm = mt / tn;
  const int tc = mt % tn;
  if (tm >= (M >> 6)) return;
  const int half = lane >> 4;
  const int r16  = lane & 15;
  const size_t m0 = (size_t)tm * 64, n0 = (size_t)tc * 32;
  const signed char* ar[4];
#pragma unroll
  for (int i = 0; i < 4; ++i) ar[i] = Aq + (m0 + i * 16 + r16) * (size_t)K;
  const signed char* br0 = Wq + (n0 + r16) * (size_t)K;
  const signed char* br1 = Wq + (n0 + 16 + r16) * (size_t)K;
  v8i acc[4][2] = {};
  for (int k0 = 0; k0 < K; k0 += 64) {
    v8i b0 = ld_bfrag_i8(br0, k0, half);
    v8i b1 = ld_bfrag_i8(br1, k0, half);
#pragma unroll
    for (int i = 0; i < 4; ++i) {
      v8i a = ld_afrag_i8(ar[i], k0, half);
      acc[i][0] = __builtin_amdgcn_wmma_i32_16x16x64_iu8(true, a, true, b0, acc[i][0], false, false);
      acc[i][1] = __builtin_amdgcn_wmma_i32_16x16x64_iu8(true, a, true, b1, acc[i][1], false, false);
    }
  }
  const float w = wsv[0];
#pragma unroll
  for (int i = 0; i < 4; ++i)
#pragma unroll
    for (int j = 0; j < 2; ++j) {
      float* crow = C + (m0 + i * 16 + half * 8) * (size_t)N + n0 + j * 16 + r16;
#pragma unroll
      for (int r = 0; r < 8; ++r) {
        size_t m = m0 + i * 16 + half * 8 + r;
        crow[(size_t)r * N] = (float)acc[i][j][r] / (xs[m] * w);
      }
    }
}

__global__ void k_conv_silu(const float* __restrict__ xz, const float* __restrict__ cw,
                            const float* __restrict__ cb, float* __restrict__ xif,
                            unsigned short* __restrict__ xib) {
  size_t i = (size_t)blockIdx.x * 256 + threadIdx.x;
  if (i >= (size_t)NTOK * DINNER) return;
  int d = (int)(i % DINNER);
  size_t tok = i / DINNER;
  int t = (int)(tok % SEQ);
  size_t brow = tok - t;
  float acc = cb[d];
#pragma unroll
  for (int j = 0; j < 4; ++j) {
    int tt = t - 3 + j;
    if (tt >= 0) acc += xz[(brow + tt) * (size_t)(2 * DINNER) + d] * cw[d * 4 + j];
  }
  float s = acc / (1.f + __expf(-acc));
  xif[i] = s;
  xib[i] = f2bf(s);
}

__global__ void k_dt_bf16(const float* __restrict__ dbc, unsigned short* __restrict__ dtb) {
  int i = blockIdx.x * 256 + threadIdx.x;
  if (i >= NTOK * DTRANK) return;
  int tok = i / DTRANK, c = i % DTRANK;
  dtb[i] = f2bf(dbc[(size_t)tok * 96 + c]);
}

__global__ void k_delta_softplus(float* __restrict__ dd, const float* __restrict__ bias) {
  size_t i = (size_t)blockIdx.x * 256 + threadIdx.x;
  if (i >= (size_t)NTOK * DINNER) return;
  float v = dd[i] + bias[i % DINNER];
  dd[i] = (v > 20.f) ? v : log1pf(__expf(v));
}

__global__ void k_scan(const float* __restrict__ dbc, const float* __restrict__ delta,
                       const float* __restrict__ xi, const unsigned char* __restrict__ reset,
                       const float* __restrict__ A_log, float* __restrict__ yscan) {
  const int b = blockIdx.x;
  const int d = blockIdx.y * 256 + threadIdx.x;
  __shared__ __align__(16) float Buf[2][64][32];

  auto issue_chunk = [&](int tck, int s) {
    float* lbase = &Buf[s][0][0];
    for (int j = threadIdx.x; j < 512; j += 256) {
      int tl = j >> 3, c4 = j & 7;
      const float* g = dbc + (size_t)(b * SEQ + tck * 64 + tl) * 96 + 64 + c4 * 4;
      unsigned loff = (unsigned)(size_t)(lbase + (size_t)tl * 32 + c4 * 4);
      asm volatile("global_load_async_to_lds_b128 %0, %1, off"
                   :: "v"(loff), "v"(g) : "memory");
    }
  };

  float Ad[DSTATE], h[DSTATE];
#pragma unroll
  for (int n = 0; n < DSTATE; ++n) { Ad[n] = -__expf(A_log[(size_t)d * DSTATE + n]); h[n] = 0.f; }

  issue_chunk(0, 0);
  for (int tck = 0; tck < SEQ / 64; ++tck) {
    const int s = tck & 1;
    asm volatile("s_wait_asynccnt 0" ::: "memory");
    __syncthreads();
    if (tck + 1 < SEQ / 64) issue_chunk(tck + 1, s ^ 1);
    for (int tl = 0; tl < 64; ++tl) {
      size_t t = (size_t)b * SEQ + tck * 64 + tl;
      if (reset[t]) {
#pragma unroll
        for (int n = 0; n < DSTATE; ++n) h[n] = 0.f;
      }
      float dv = delta[t * DINNER + d];
      float xv = xi[t * DINNER + d];
      float y = 0.f;
#pragma unroll
      for (int n = 0; n < DSTATE; ++n) {
        h[n] = h[n] * __expf(dv * Ad[n]) + dv * Buf[s][tl][n] * xv;
        y += h[n] * Buf[s][tl][16 + n];
      }
      yscan[t * DINNER + d] = y;
    }
    __syncthreads();
  }
}

__global__ void k_y_gate(const float* __restrict__ ys, const float* __restrict__ xif,
                         const float* __restrict__ xz, const float* __restrict__ Dp,
                         unsigned short* __restrict__ ybf) {
  size_t i = (size_t)blockIdx.x * 256 + threadIdx.x;
  if (i >= (size_t)NTOK * DINNER) return;
  int d = (int)(i % DINNER);
  size_t tok = i / DINNER;
  float y = ys[i] + xif[i] * Dp[d];
  float z = xz[tok * (size_t)(2 * DINNER) + DINNER + d];
  float g = z / (1.f + __expf(-z));
  ybf[i] = f2bf(y * g);
}

__global__ void k_norm1_quant(const float* __restrict__ x, const float* __restrict__ m,
                              const float* __restrict__ nw, float* __restrict__ x1,
                              signed char* __restrict__ xq, float* __restrict__ xs) {
  int tok = blockIdx.x;
  __shared__ float red[256];
  float s[4], v[4], ss = 0.f;
#pragma unroll
  for (int r = 0; r < 4; ++r) {
    int j = threadIdx.x + r * 256;
    s[r] = x[(size_t)tok * DMODEL + j] + m[(size_t)tok * DMODEL + j];
    ss += s[r] * s[r];
  }
  red[threadIdx.x] = ss; __syncthreads();
  for (int o = 128; o > 0; o >>= 1) { if (threadIdx.x < o) red[threadIdx.x] += red[threadIdx.x + o]; __syncthreads(); }
  float rms = rsqrtf(red[0] / (float)DMODEL + 1e-6f);
  __syncthreads();
  float amax = 0.f;
#pragma unroll
  for (int r = 0; r < 4; ++r) {
    int j = threadIdx.x + r * 256;
    v[r] = s[r] * rms * nw[j];
    x1[(size_t)tok * DMODEL + j] = v[r];
    amax = fmaxf(amax, fabsf(v[r]));
  }
  red[threadIdx.x] = amax; __syncthreads();
  for (int o = 128; o > 0; o >>= 1) { if (threadIdx.x < o) red[threadIdx.x] = fmaxf(red[threadIdx.x], red[threadIdx.x + o]); __syncthreads(); }
  float sc = 127.f / fmaxf(red[0], 1e-5f);
  if (threadIdx.x == 0) xs[tok] = sc;
#pragma unroll
  for (int r = 0; r < 4; ++r) {
    int j = threadIdx.x + r * 256;
    float q = fminf(fmaxf(rintf(v[r] * sc), -128.f), 127.f);
    xq[(size_t)tok * DMODEL + j] = (signed char)q;
  }
}

__global__ void k_abs_partial(const float* __restrict__ w, int n, float* __restrict__ partial) {
  __shared__ float red[256];
  float s = 0.f;
  for (size_t i = (size_t)blockIdx.x * 256 + threadIdx.x; i < (size_t)n; i += 256 * 256)
    s += fabsf(w[i]);
  red[threadIdx.x] = s; __syncthreads();
  for (int o = 128; o > 0; o >>= 1) { if (threadIdx.x < o) red[threadIdx.x] += red[threadIdx.x + o]; __syncthreads(); }
  if (threadIdx.x == 0) partial[blockIdx.x] = red[0];
}

__global__ void k_ws_final(const float* __restrict__ partials, float* __restrict__ wsout) {
  __shared__ float red[256];
  for (int mtx = 0; mtx < 3; ++mtx) {
    red[threadIdx.x] = partials[mtx * 256 + threadIdx.x];
    __syncthreads();
    for (int o = 128; o > 0; o >>= 1) { if (threadIdx.x < o) red[threadIdx.x] += red[threadIdx.x + o]; __syncthreads(); }
    if (threadIdx.x == 0) {
      float mean = red[0] / 4194304.0f;
      wsout[mtx] = 1.0f / fmaxf(mean, 1e-5f);
    }
    __syncthreads();
  }
}

__global__ void k_wquant(const float* __restrict__ w, const float* __restrict__ wsv,
                         signed char* __restrict__ wq, int n) {
  int i = blockIdx.x * 256 + threadIdx.x;
  if (i < n) {
    float q = fminf(fmaxf(rintf(w[i] * wsv[0]), -1.f), 1.f);
    wq[i] = (signed char)q;
  }
}

__global__ void k_gateup_quant(const float* __restrict__ gf, const float* __restrict__ uf,
                               signed char* __restrict__ pq, float* __restrict__ xs2) {
  int tok = blockIdx.x;
  __shared__ float red[256];
  float p[16], amax = 0.f;
#pragma unroll
  for (int r = 0; r < 16; ++r) {
    size_t j = (size_t)tok * DFF + threadIdx.x + r * 256;
    float g = gf[j];
    p[r] = uf[j] / (1.f + __expf(-g));
    amax = fmaxf(amax, fabsf(p[r]));
  }
  red[threadIdx.x] = amax; __syncthreads();
  for (int o = 128; o > 0; o >>= 1) { if (threadIdx.x < o) red[threadIdx.x] = fmaxf(red[threadIdx.x], red[threadIdx.x + o]); __syncthreads(); }
  float sc = 127.f / fmaxf(red[0], 1e-5f);
  if (threadIdx.x == 0) xs2[tok] = sc;
#pragma unroll
  for (int r = 0; r < 16; ++r) {
    size_t j = (size_t)tok * DFF + threadIdx.x + r * 256;
    float q = fminf(fmaxf(rintf(p[r] * sc), -128.f), 127.f);
    pq[j] = (signed char)q;
  }
}

__global__ void k_norm2(const float* __restrict__ x1, const float* __restrict__ f,
                        const float* __restrict__ nw, float* __restrict__ out) {
  int tok = blockIdx.x;
  __shared__ float red[256];
  float s[4], ss = 0.f;
#pragma unroll
  for (int r = 0; r < 4; ++r) {
    int j = threadIdx.x + r * 256;
    s[r] = x1[(size_t)tok * DMODEL + j] + f[(size_t)tok * DMODEL + j];
    ss += s[r] * s[r];
  }
  red[threadIdx.x] = ss; __syncthreads();
  for (int o = 128; o > 0; o >>= 1) { if (threadIdx.x < o) red[threadIdx.x] += red[threadIdx.x + o]; __syncthreads(); }
  float rms = rsqrtf(red[0] / (float)DMODEL + 1e-6f);
#pragma unroll
  for (int r = 0; r < 4; ++r) {
    int j = threadIdx.x + r * 256;
    out[(size_t)tok * DMODEL + j] = s[r] * rms * nw[j];
  }
}

extern "C" void kernel_launch(void* const* d_in, const int* in_sizes, int n_in,
                              void* d_out, int out_size, void* d_ws, size_t ws_size,
                              hipStream_t stream) {
  (void)in_sizes; (void)n_in; (void)out_size; (void)ws_size;
  const float* x          = (const float*)d_in[0];
  const unsigned char* rm = (const unsigned char*)d_in[1];
  const float* in_proj_w  = (const float*)d_in[2];
  const float* conv_w     = (const float*)d_in[3];
  const float* conv_b     = (const float*)d_in[4];
  const float* x_proj_w   = (const float*)d_in[5];
  const float* dt_proj_w  = (const float*)d_in[6];
  const float* dt_proj_b  = (const float*)d_in[7];
  const float* A_log      = (const float*)d_in[8];
  const float* Dvec       = (const float*)d_in[9];
  const float* out_proj_w = (const float*)d_in[10];
  const float* norm1_w    = (const float*)d_in[11];
  const float* norm2_w    = (const float*)d_in[12];
  const float* gate_w     = (const float*)d_in[13];
  const float* up_w       = (const float*)d_in[14];
  const float* down_w     = (const float*)d_in[15];
  float* out = (float*)d_out;

  unsigned char* base = (unsigned char*)d_ws;
  size_t off = 0;
  auto alloc = [&](size_t bytes) -> void* {
    void* p = base + off; off += (bytes + 255) & ~(size_t)255; return p;
  };
  unsigned short* XB   = (unsigned short*)alloc((size_t)NTOK * DMODEL * 2);
  unsigned short* WIN  = (unsigned short*)alloc((size_t)2 * DINNER * DMODEL * 2);
  unsigned short* WXP  = (unsigned short*)alloc((size_t)96 * DINNER * 2);
  unsigned short* WDT  = (unsigned short*)alloc((size_t)DINNER * DTRANK * 2);
  unsigned short* WOUT = (unsigned short*)alloc((size_t)DMODEL * DINNER * 2);
  float*          XZ   = (float*)alloc((size_t)NTOK * 2 * DINNER * 4);
  float*          XIF  = (float*)alloc((size_t)NTOK * DINNER * 4);
  unsigned short* XIB  = (unsigned short*)alloc((size_t)NTOK * DINNER * 2);
  float*          DBC  = (float*)alloc((size_t)NTOK * 96 * 4);
  unsigned short* DTB  = (unsigned short*)alloc((size_t)NTOK * DTRANK * 2);
  float*          DELTA= (float*)alloc((size_t)NTOK * DINNER * 4);
  float*          YSC  = (float*)alloc((size_t)NTOK * DINNER * 4);
  unsigned short* YBF  = (unsigned short*)alloc((size_t)NTOK * DINNER * 2);
  float*          MB   = (float*)alloc((size_t)NTOK * DMODEL * 4);
  float*          X1   = (float*)alloc((size_t)NTOK * DMODEL * 4);
  signed char*    XQ1  = (signed char*)alloc((size_t)NTOK * DMODEL);
  float*          XS1  = (float*)alloc((size_t)NTOK * 4);
  float*          XS2  = (float*)alloc((size_t)NTOK * 4);
  float*          WSP  = (float*)alloc(3 * 256 * 4);
  float*          WSS  = (float*)alloc(256);
  signed char*    WQG  = (signed char*)alloc((size_t)DFF * DMODEL);
  signed char*    WQU  = (signed char*)alloc((size_t)DFF * DMODEL);
  signed char*    WQD  = (signed char*)alloc((size_t)DMODEL * DFF);
  float*       GATEF = XZ;
  float*       UPF   = DELTA;
  signed char* PQ    = (signed char*)XIF;
  float*       FBUF  = (float*)YBF;

  const int TB = 256;
  auto cd = [](size_t a, size_t b) -> unsigned { return (unsigned)((a + b - 1) / b); };
  auto gemm_bf16 = [&](const unsigned short* A, const unsigned short* W, float* C,
                       int M, int N, int K) {
    int tiles = (M / 64) * (N / 32);
    k_gemm_bf16_blk<<<dim3(tiles / 4), dim3(128), 0, stream>>>(A, W, C, M, N, K);
  };
  auto gemm_iu8 = [&](const signed char* A, const signed char* W, const float* xs,
                      const float* ws, float* C, int M, int N, int K) {
    int tiles = (M / 64) * (N / 32);
    k_gemm_iu8_blk<<<dim3(tiles / 4), dim3(128), 0, stream>>>(A, W, xs, ws, C, M, N, K);
  };

  k_f32_to_bf16<<<cd((size_t)2*DINNER*DMODEL, TB), TB, 0, stream>>>(in_proj_w, WIN, 2*DINNER*DMODEL);
  k_f32_to_bf16<<<cd((size_t)96*DINNER, TB), TB, 0, stream>>>(x_proj_w, WXP, 96*DINNER);
  k_f32_to_bf16<<<cd((size_t)DINNER*DTRANK, TB), TB, 0, stream>>>(dt_proj_w, WDT, DINNER*DTRANK);
  k_f32_to_bf16<<<cd((size_t)DMODEL*DINNER, TB), TB, 0, stream>>>(out_proj_w, WOUT, DMODEL*DINNER);
  k_f32_to_bf16<<<cd((size_t)NTOK*DMODEL, TB), TB, 0, stream>>>(x, XB, NTOK*DMODEL);

  gemm_bf16(XB, WIN, XZ, NTOK, 2 * DINNER, DMODEL);
  k_conv_silu<<<cd((size_t)NTOK*DINNER, TB), TB, 0, stream>>>(XZ, conv_w, conv_b, XIF, XIB);
  gemm_bf16(XIB, WXP, DBC, NTOK, 96, DINNER);
  k_dt_bf16<<<cd((size_t)NTOK*DTRANK, TB), TB, 0, stream>>>(DBC, DTB);
  gemm_bf16(DTB, WDT, DELTA, NTOK, DINNER, DTRANK);
  k_delta_softplus<<<cd((size_t)NTOK*DINNER, TB), TB, 0, stream>>>(DELTA, dt_proj_b);
  k_scan<<<dim3(BATCH, DINNER / 256), dim3(256), 0, stream>>>(DBC, DELTA, XIF, rm, A_log, YSC);
  k_y_gate<<<cd((size_t)NTOK*DINNER, TB), TB, 0, stream>>>(YSC, XIF, XZ, Dvec, YBF);
  gemm_bf16(YBF, WOUT, MB, NTOK, DMODEL, DINNER);
  k_norm1_quant<<<dim3(NTOK), dim3(256), 0, stream>>>(x, MB, norm1_w, X1, XQ1, XS1);

  k_abs_partial<<<dim3(256), dim3(256), 0, stream>>>(gate_w, DFF * DMODEL, WSP);
  k_abs_partial<<<dim3(256), dim3(256), 0, stream>>>(up_w,   DFF * DMODEL, WSP + 256);
  k_abs_partial<<<dim3(256), dim3(256), 0, stream>>>(down_w, DMODEL * DFF, WSP + 512);
  k_ws_final<<<dim3(1), dim3(256), 0, stream>>>(WSP, WSS);
  k_wquant<<<cd((size_t)DFF*DMODEL, TB), TB, 0, stream>>>(gate_w, WSS + 0, WQG, DFF * DMODEL);
  k_wquant<<<cd((size_t)DFF*DMODEL, TB), TB, 0, stream>>>(up_w,   WSS + 1, WQU, DFF * DMODEL);
  k_wquant<<<cd((size_t)DMODEL*DFF, TB), TB, 0, stream>>>(down_w, WSS + 2, WQD, DMODEL * DFF);

  gemm_iu8(XQ1, WQG, XS1, WSS + 0, GATEF, NTOK, DFF, DMODEL);
  gemm_iu8(XQ1, WQU, XS1, WSS + 1, UPF,   NTOK, DFF, DMODEL);
  k_gateup_quant<<<dim3(NTOK), dim3(256), 0, stream>>>(GATEF, UPF, PQ, XS2);
  gemm_iu8(PQ, WQD, XS2, WSS + 2, FBUF, NTOK, DMODEL, DFF);
  k_norm2<<<dim3(NTOK), dim3(256), 0, stream>>>(X1, FBUF, norm2_w, out);
}
